// MPNN_67095979098840
// MI455X (gfx1250) — matrix operand order repaired and checked by reading, not compiled, not run on hardware
//
#include <hip/hip_runtime.h>
#include <hip/hip_bf16.h>

#define NN 1024
#define NODE_DIM 128
#define EDGE_DIM 16
#define MSG_DIM 64
#define IN_DIM 272   // 2*NODE_DIM + EDGE_DIM

typedef _Float16     h16v __attribute__((ext_vector_type(16)));
typedef float        f8v  __attribute__((ext_vector_type(8)));
typedef unsigned int u8v  __attribute__((ext_vector_type(8)));

// ---------------------------------------------------------------------------
// Projection: hi = H @ Wi^T, hjb = H @ Wj^T + b1, and zero M.
// One thread per (node, msg-col). 1024*64 = 65536 threads.
// ---------------------------------------------------------------------------
__global__ __launch_bounds__(256) void proj_kernel(
    const float* __restrict__ H, const float* __restrict__ W1,
    const float* __restrict__ b1,
    float* __restrict__ hi, float* __restrict__ hjb, float* __restrict__ M)
{
  int idx = blockIdx.x * 256 + threadIdx.x;     // 0..65535
  int i = idx >> 6;
  int m = idx & 63;
  const float* h  = H  + (size_t)i * NODE_DIM;
  const float* wi = W1 + (size_t)m * IN_DIM;          // cols 0..127
  const float* wj = wi + NODE_DIM;                    // cols 128..255
  float s0 = 0.f, s1 = 0.f;
#pragma unroll 8
  for (int k = 0; k < NODE_DIM; ++k) {
    float hv = h[k];
    s0 += hv * wi[k];
    s1 += hv * wj[k];
  }
  hi[idx]  = s0;
  hjb[idx] = s1 + b1[m];
  M[idx]   = 0.f;       // zero accumulator for message reduction
}

// ---------------------------------------------------------------------------
// Fused message kernel (transposed formulation, no LDS round-trip).
// Block = 4 waves; each wave owns the block's 16-row i-tile and 32 j's.
// Per j:
//   pre^T (64x16) = We(64x16,K-padded) @ E_row^T(16x16)  + hjb (K=16 fold)
//                   + hi (WMMA C operand, loop-invariant)      [4x wmma]
//   cvt_pk then packed-f16 relu  -> A-frags of GEMM2 directly (layout id.)
//   msg (16x64)   = reluPre @ W2^T                             [8x wmma]
//   macc += A[i,j]*msg ; deg += A[i,j]        (b2 folded at flush via deg)
// Flush: global_atomic_add_f32 into M.
// ---------------------------------------------------------------------------
__global__ __launch_bounds__(128) void msg_kernel(
    const float* __restrict__ E, const int* __restrict__ A,
    const float* __restrict__ W1, const float* __restrict__ W2,
    const float* __restrict__ b2,
    const float* __restrict__ hi, const float* __restrict__ hjb,
    float* __restrict__ M)
{
  const int lane  = threadIdx.x & 31;
  const int wave  = threadIdx.x >> 5;
  const int nlo   = lane & 15;        // i_local (GEMM1 B col / GEMM2 A row)
  const int hhalf = lane >> 4;        // 0: lanes 0-15, 1: lanes 16-31
  const int i0    = blockIdx.x * 16;
  const int jbase = blockIdx.y * 128 + wave * 32;   // 32 j's per wave

  // --- A fragments for We (A1 = We, rows = msg features, K=16 padded 32) ---
  // A layout: lanes0-15 halves0-7 = K0..7, lanes16-31 halves0-7 = K8..15;
  // half 8 (K=16, lanes<16) is the per-j hjb fold slot; rest zero.
  h16v aWe[4];
#pragma unroll
  for (int mt = 0; mt < 4; ++mt) {
    int m = mt * 16 + nlo;
    const float* src = W1 + (size_t)m * IN_DIM + 2 * NODE_DIM + hhalf * 8;
    h16v a;
#pragma unroll
    for (int t = 0; t < 8; ++t) a[t] = (_Float16)src[t];
#pragma unroll
    for (int t = 8; t < 16; ++t) a[t] = (_Float16)0.f;
    aWe[mt] = a;
  }

  // --- B fragments for W2 (GEMM2: B = W2^T, 64x64): [nt][ks] ---------------
  // B layout: lanes0-15 halves = K 0..7,16..23 of col n; lanes16-31 = K 8..15,24..31.
  h16v bW2[4][2];
#pragma unroll
  for (int nt = 0; nt < 4; ++nt) {
    int n = nt * 16 + nlo;
#pragma unroll
    for (int ks = 0; ks < 2; ++ks) {
      int kb = ks * 32 + hhalf * 8;
      const float* src = W2 + (size_t)n * MSG_DIM + kb;
      h16v b;
#pragma unroll
      for (int t = 0; t < 16; ++t) b[t] = (_Float16)src[t < 8 ? t : t + 8];
      bW2[nt][ks] = b;
    }
  }

  // constant B1 frame for lanes<16: K16 row = 1.0 (hjb fold), K17..31 = 0
  h16v bOne;
#pragma unroll
  for (int t = 0; t < 16; ++t) bOne[t] = (_Float16)0.f;
  bOne[0] = (_Float16)1.f;
  const u8v uOne = __builtin_bit_cast(u8v, bOne);

  // zero vector for packed-f16 relu
  h16v hzero;
#pragma unroll
  for (int t = 0; t < 16; ++t) hzero[t] = (_Float16)0.f;

  // --- loop-invariant hi tile as WMMA C operand ----------------------------
  // pre^T element [m, i]: m = mt*16 + v + 8*hhalf, i = i0 + nlo.
  f8v hbase[4];
#pragma unroll
  for (int mt = 0; mt < 4; ++mt) {
    const float* src = hi + (size_t)(i0 + nlo) * MSG_DIM + mt * 16 + hhalf * 8;
#pragma unroll
    for (int v = 0; v < 8; ++v) hbase[mt][v] = src[v];
  }

  // register accumulators: 16x64 partial of M, plus per-row degree
  f8v macc[4];
#pragma unroll
  for (int nt = 0; nt < 4; ++nt)
#pragma unroll
    for (int v = 0; v < 8; ++v) macc[nt][v] = 0.f;
  float dg[8];
#pragma unroll
  for (int v = 0; v < 8; ++v) dg[v] = 0.f;

  for (int jj = 0; jj < 32; ++jj) {
    const int j = jbase + jj;

    // prefetch E rows a few j ahead (streaming)
    {
      int jp = (j + 8 < NN) ? j + 8 : j;
      __builtin_prefetch(E + ((size_t)(i0 + nlo) * NN + jp) * EDGE_DIM, 0, 1);
    }

    // --- B1 = E tile transposed (K=16 edge dims padded to 32, N=16 i-rows) -
    // lanes0-15: halves0-7 = K0..7 = E[i0+nlo, j, 0..7], half 8 = 1.0 (K16, hjb fold);
    // lanes16-31: halves0-7 = K8..15 = E[i0+nlo, j, 8..15]; rest 0.
    h16v bE;
    {
      const float4* ep = reinterpret_cast<const float4*>(
          E + ((size_t)(i0 + nlo) * NN + j) * EDGE_DIM);
      float4 q0 = ep[0], q1 = ep[1], q2 = ep[2], q3 = ep[3];
      float ev[16] = {q0.x, q0.y, q0.z, q0.w, q1.x, q1.y, q1.z, q1.w,
                      q2.x, q2.y, q2.z, q2.w, q3.x, q3.y, q3.z, q3.w};
      h16v beRaw;
#pragma unroll
      for (int t = 0; t < 16; ++t) beRaw[t] = (_Float16)ev[t];
      u8v ue = __builtin_bit_cast(u8v, beRaw);
      u8v ub;
#pragma unroll
      for (int k = 0; k < 8; ++k) ub[k] = k < 4 ? ue[k + 4 * hhalf] : ((k == 4 && !hhalf) ? uOne[0] : 0u);
      bE = __builtin_bit_cast(h16v, ub);
    }

    // --- fold hjb[j, m] into A1 at K=16 (lanes<16, half 8) ----------------
#pragma unroll
    for (int mt = 0; mt < 4; ++mt) {
      float hv = hjb[(size_t)j * MSG_DIM + mt * 16 + nlo];
      aWe[mt][8] = hhalf ? (_Float16)0.f : (_Float16)hv;
    }

    // adjacency mask for my 8 C-layout rows
    float amask[8];
#pragma unroll
    for (int v = 0; v < 8; ++v) {
      int row = v + 8 * hhalf;
      amask[v] = (float)A[(size_t)(i0 + row) * NN + j];
    }

    // --- GEMM1: pre^T = We @ E^T + hjb(K-fold) + hi(C operand) ------------
    f8v p[4];
#pragma unroll
    for (int mt = 0; mt < 4; ++mt)
      p[mt] = __builtin_amdgcn_wmma_f32_16x16x32_f16(
          false, aWe[mt], false, bE, (short)0, hbase[mt], false, false);

    // --- pack then relu in packed f16 (cvt is monotonic, cvt(0)=0, so
    //     relu(cvt(x)) == cvt(relu(x)); lowers to v_pk_max_num_f16) --------
    h16v aM0, aM1;
#pragma unroll
    for (int t = 0; t < 8; ++t) {
      aM0[t]     = (_Float16)p[0][t];
      aM0[8 + t] = (_Float16)p[1][t];
      aM1[t]     = (_Float16)p[2][t];
      aM1[8 + t] = (_Float16)p[3][t];
    }
    aM0 = __builtin_elementwise_max(aM0, hzero);
    aM1 = __builtin_elementwise_max(aM1, hzero);

    // --- GEMM2: msg = relu(pre) @ W2^T ; mask-accumulate ------------------
#pragma unroll
    for (int nt = 0; nt < 4; ++nt) {
      f8v c;
#pragma unroll
      for (int v = 0; v < 8; ++v) c[v] = 0.f;
      c = __builtin_amdgcn_wmma_f32_16x16x32_f16(false, aM0, false,
                                                 bW2[nt][0], (short)0, c,
                                                 false, false);
      c = __builtin_amdgcn_wmma_f32_16x16x32_f16(false, aM1, false,
                                                 bW2[nt][1], (short)0, c,
                                                 false, false);
#pragma unroll
      for (int v = 0; v < 8; ++v)
        macc[nt][v] += amask[v] * c[v];
    }
#pragma unroll
    for (int v = 0; v < 8; ++v) dg[v] += amask[v];
  }

  // --- flush partials: M[i, n] += macc + b2[n]*deg(i) ----------------------
#pragma unroll
  for (int nt = 0; nt < 4; ++nt) {
    int n = nt * 16 + nlo;
    float b2n = b2[n];
#pragma unroll
    for (int v = 0; v < 8; ++v) {
      int row = v + 8 * hhalf;
      atomicAdd(&M[(size_t)(i0 + row) * MSG_DIM + n],
                macc[nt][v] + b2n * dg[v]);
    }
  }
}

// ---------------------------------------------------------------------------
// GRU update: one thread per (node, hidden-col). 1024*128 threads.
// ---------------------------------------------------------------------------
__global__ __launch_bounds__(256) void gru_kernel(
    const float* __restrict__ Hin, const float* __restrict__ Mg,
    const float* __restrict__ W_ih, const float* __restrict__ b_ih,
    const float* __restrict__ W_hh, const float* __restrict__ b_hh,
    float* __restrict__ Hout)
{
  int idx = blockIdx.x * 256 + threadIdx.x;   // 0..131071
  int i = idx >> 7;
  int c = idx & 127;
  const float* m = Mg  + (size_t)i * MSG_DIM;
  const float* h = Hin + (size_t)i * NODE_DIM;

  float gi[3], gh[3];
#pragma unroll
  for (int g = 0; g < 3; ++g) {
    int rowc = g * NODE_DIM + c;
    const float* wih = W_ih + (size_t)rowc * MSG_DIM;
    float s = b_ih[rowc];
#pragma unroll 8
    for (int k = 0; k < MSG_DIM; ++k) s += m[k] * wih[k];
    gi[g] = s;
    const float* whh = W_hh + (size_t)rowc * NODE_DIM;
    float t = b_hh[rowc];
#pragma unroll 8
    for (int k = 0; k < NODE_DIM; ++k) t += h[k] * whh[k];
    gh[g] = t;
  }
  float r = 1.f / (1.f + __expf(-(gi[0] + gh[0])));
  float z = 1.f / (1.f + __expf(-(gi[1] + gh[1])));
  float n = tanhf(gi[2] + r * gh[2]);
  Hout[idx] = (1.f - z) * n + z * h[c];
}

// ---------------------------------------------------------------------------
extern "C" void kernel_launch(void* const* d_in, const int* in_sizes, int n_in,
                              void* d_out, int out_size, void* d_ws,
                              size_t ws_size, hipStream_t stream)
{
  const float* X    = (const float*)d_in[0];
  const int*   A    = (const int*)  d_in[1];
  const float* E    = (const float*)d_in[2];
  const float* W1   = (const float*)d_in[3];
  const float* b1   = (const float*)d_in[4];
  const float* W2   = (const float*)d_in[5];
  const float* b2   = (const float*)d_in[6];
  const float* W_ih = (const float*)d_in[7];
  const float* b_ih = (const float*)d_in[8];
  const float* W_hh = (const float*)d_in[9];
  const float* b_hh = (const float*)d_in[10];
  float* out = (float*)d_out;

  float* hi  = (float*)d_ws;                     // 1024*64
  float* hjb = hi  + NN * MSG_DIM;               // 1024*64
  float* M   = hjb + NN * MSG_DIM;               // 1024*64
  float* H1  = M   + NN * MSG_DIM;               // 1024*128
  float* H2  = H1  + NN * NODE_DIM;              // 1024*128

  const float* Hin = X;
  float* Houts[3] = {H1, H2, out};

  for (int t = 0; t < 3; ++t) {
    proj_kernel<<<256, 256, 0, stream>>>(Hin, W1, b1, hi, hjb, M);
    dim3 g(NN / 16, 8);
    msg_kernel<<<g, 128, 0, stream>>>(E, A, W1, W2, b2, hi, hjb, M);
    gru_kernel<<<512, 256, 0, stream>>>(Hin, M, W_ih, b_ih, W_hh, b_hh,
                                        Houts[t]);
    Hin = Houts[t];
  }
}
